// GatedCrossAttention_154618822691
// MI455X (gfx1250) — matrix operand order repaired and checked by reading, not compiled, not run on hardware
//
#include <hip/hip_runtime.h>
#include <hip/hip_bf16.h>

typedef __attribute__((ext_vector_type(16))) _Float16 v16h;
typedef __attribute__((ext_vector_type(8)))  _Float16 v8h;
typedef __attribute__((ext_vector_type(8)))  float    v8f;
typedef __attribute__((ext_vector_type(4)))  unsigned int u32x4;
typedef __attribute__((ext_vector_type(4)))  int i32x4;
typedef __attribute__((ext_vector_type(8)))  int i32x8;

union Frag16 { v16h v; v8h v8[2]; _Float16 h[16]; };
union FragC  { v8f  v; float f[8]; };

// ---------------------------------------------------------------------------
// f32 -> f16 convert (plain) and transpose-convert (CxHW -> HWxC pixel-major)
// ---------------------------------------------------------------------------
__global__ void f32_to_f16_kernel(const float* __restrict__ src,
                                  _Float16* __restrict__ dst, long n) {
    long t = (long)blockIdx.x * blockDim.x + threadIdx.x;
    if (t < n) dst[t] = (_Float16)src[t];
}

__global__ void f32_to_f16_transpose_kernel(const float* __restrict__ src, // C x HW
                                            _Float16* __restrict__ dst,   // HW x C
                                            int Cc, int HWp) {
    int t = blockIdx.x * blockDim.x + threadIdx.x;
    if (t >= Cc * HWp) return;
    int p = t / Cc, c = t - (t / Cc) * Cc;
    dst[t] = (_Float16)src[(long)c * HWp + p];
}

// ---------------------------------------------------------------------------
// Weight repack to tap-major f16: dst[co][ (ky*KS+kx)*Cin + ci ]
// src is OIHW (swap_io=0) or IOHW (swap_io=1), optionally spatially flipped.
// ---------------------------------------------------------------------------
__global__ void pack_tapmajor_kernel(const float* __restrict__ src,
                                     _Float16* __restrict__ dst,
                                     int Cout, int Cin, int KS,
                                     int swap_io, int flip) {
    int t = blockIdx.x * blockDim.x + threadIdx.x;
    const int KK = KS * KS;
    int total = Cout * Cin * KK;
    if (t >= total) return;
    int co  = t / (Cin * KK);
    int rem = t - co * Cin * KK;
    int tap = rem / Cin;
    int ci  = rem - tap * Cin;
    int ky = tap / KS, kx = tap - (tap / KS) * KS;
    if (flip) { ky = KS - 1 - ky; kx = KS - 1 - kx; }
    long si = swap_io ? ((long)ci * Cout + co) : ((long)co * Cin + ci);
    dst[t] = (_Float16)src[si * KK + ky * KS + kx];
}

// ---------------------------------------------------------------------------
// Implicit-GEMM conv via WMMA.
//   X: HWin x Cin  (pixel-major f16)      -> A frags are 2x global_load_b128
//   W: Cout x (KK*Cin) tap-major, TDM-staged into LDS -> B frags 2x ds_load_b128
//   Y: HWout x Cout (pixel-major f16)
// Border/dilation handled by a per-tap pointer select onto a zero row.
// Block: one N-tile (16 Cout) shared by 8 waves covering 8 M-tiles.
// ---------------------------------------------------------------------------
__global__ void __launch_bounds__(256)
conv_implicit_wmma(const _Float16* __restrict__ X,
                   const _Float16* __restrict__ W,
                   const _Float16* __restrict__ zrow,
                   const float* __restrict__ bias,
                   _Float16* __restrict__ Y,
                   int Cin, int Hin, int Win,
                   int Hout, int Wout, int Cout,
                   int KS, int pad, int dil, int relu) {
    extern __shared__ _Float16 wlds[];           // 16 x KT weight tile
    const int lane = threadIdx.x & 31;
    const int wid  = threadIdx.x >> 5;
    const int Mtiles = (Hout * Wout) >> 4;
    const int Ntiles = Cout >> 4;
    const int KK = KS * KS;
    const int KT = KK * Cin;

    const int nt  = blockIdx.x % Ntiles;
    const int mtb = (blockIdx.x / Ntiles) * 8;
    int mt = mtb + wid;
    if (mt >= Mtiles) mt = Mtiles - 1;           // clamp: EXEC stays all-1s

    // ---- TDM: DMA the 16 x KT f16 weight tile into LDS (once per block) ----
    if (threadIdx.x == 0) {
        unsigned long ga = (unsigned long)(W + (long)(nt << 4) * KT);
        unsigned lds_off = (unsigned)(unsigned long)&wlds[0];
        u32x4 g0;
        g0[0] = 1u;                                           // count=1, user D#
        g0[1] = lds_off;                                      // LDS byte address
        g0[2] = (unsigned)(ga & 0xffffffffu);                 // global_addr[31:0]
        g0[3] = (unsigned)((ga >> 32) & 0x01ffffffu) | (2u << 30); // addr[56:32]|type=2
        i32x8 g1;
        g1[0] = 1 << 16;                                      // data_size=1 (2B)
        g1[1] = (KT & 0xffff) << 16;                          // tensor_dim0 lo16
        g1[2] = ((KT >> 16) & 0xffff) | (16 << 16);           // dim0 hi | tensor_dim1 lo
        g1[3] = (KT & 0xffff) << 16;                          // tile_dim0 = KT
        g1[4] = 16;                                           // tile_dim1 = 16
        g1[5] = KT;                                           // tensor_dim0_stride lo32
        g1[6] = 0;
        g1[7] = 0;
        i32x4 z4 = {0, 0, 0, 0};
#if __has_include(<hip/amd_detail/amd_gfx1250_TDM.h>)
        i32x8 z8 = {0, 0, 0, 0, 0, 0, 0, 0};
        __builtin_amdgcn_tensor_load_to_lds(g0, g1, z4, z4, z8, 0);
#else
        __builtin_amdgcn_tensor_load_to_lds(g0, g1, z4, z4, 0);
#endif
        __builtin_amdgcn_s_wait_tensorcnt((short)0);
    }
    __syncthreads();

    const int l15 = lane & 15;
    const int aoff = (lane < 16) ? 0 : 16;
    const int boff = (lane < 16) ? 0 : 16;
    const int m  = (mt << 4) + l15;
    const int oy = m / Wout, ox = m - (m / Wout) * Wout;
    const int n  = (nt << 4) + l15;

    FragC acc;
    const float bv = bias[n];
#pragma unroll
    for (int i = 0; i < 8; ++i) acc.f[i] = bv;

    const _Float16* wl = wlds + (long)l15 * KT;

    for (int tap = 0; tap < KK; ++tap) {
        const int ky = tap / KS, kx = tap - (tap / KS) * KS;
        const int ty = oy + ky - pad, tx = ox + kx - pad;
        int iy = ty, ix = tx;
        int valid = (ty >= 0) & (tx >= 0);
        if (dil == 2) { valid &= !((ty | tx) & 1); iy = ty >> 1; ix = tx >> 1; }
        valid &= (iy < Hin) & (ix < Win);
        // single pointer select per tap; invalid taps read a zero row
        const _Float16* xr = valid ? (X + ((long)iy * Win + ix) * Cin) : zrow;
        const _Float16* wt = wl + tap * Cin + boff;

        for (int cb = 0; cb < Cin; cb += 32) {
            Frag16 a, b;
            a.v8[0] = *(const v8h*)(xr + cb + aoff);          // global_load_b128
            a.v8[1] = *(const v8h*)(xr + cb + aoff + 8);
            b.v8[0] = *(const v8h*)(wt + cb);                 // ds_load_b128
            b.v8[1] = *(const v8h*)(wt + cb + 8);
            acc.v = __builtin_amdgcn_wmma_f32_16x16x32_f16(
                false, a.v, false, b.v, (short)0, acc.v, false, false);
        }
    }

    const int rbase = (lane < 16) ? 0 : 8;       // D-layout rows for this lane
#pragma unroll
    for (int r = 0; r < 8; ++r) {
        float v = acc.f[r];
        if (relu) v = fmaxf(v, 0.0f);
        Y[(long)((mt << 4) + rbase + r) * Cout + n] = (_Float16)v;  // pixel-major
    }
}

// ---------------------------------------------------------------------------
// Row-major A (MxK) x row-major B (NxK, used transposed) -> C (MxN f16), WMMA
// ---------------------------------------------------------------------------
__global__ void __launch_bounds__(256)
gemm_rcr_wmma(const _Float16* __restrict__ A,
              const _Float16* __restrict__ B,
              const float* __restrict__ bias,
              _Float16* __restrict__ C,
              int M, int N, int K) {
    const int lane = threadIdx.x & 31;
    const int wid  = threadIdx.x >> 5;
    const int Mtiles = M >> 4, Ntiles = N >> 4;
    const int gw = blockIdx.x * (blockDim.x >> 5) + wid;
    if (gw >= Mtiles * Ntiles) return;
    const int mt = gw % Mtiles, nt = gw / Mtiles;
    const int l15 = lane & 15;
    const int aoff = (lane < 16) ? 0 : 16;
    const int boff = (lane < 16) ? 0 : 16;
    const _Float16* arow = A + (long)((mt << 4) + l15) * K;
    const _Float16* brow = B + (long)((nt << 4) + l15) * K;

    FragC acc;
    const float bv = bias[(nt << 4) + l15];
#pragma unroll
    for (int i = 0; i < 8; ++i) acc.f[i] = bv;

    for (int kb = 0; kb < K; kb += 32) {
        Frag16 a, b;
        a.v8[0] = *(const v8h*)(arow + kb + aoff);
        a.v8[1] = *(const v8h*)(arow + kb + aoff + 8);
        b.v8[0] = *(const v8h*)(brow + kb + boff);
        b.v8[1] = *(const v8h*)(brow + kb + boff + 8);
        __builtin_prefetch((const void*)(brow + kb + 64), 0, 0);
        acc.v = __builtin_amdgcn_wmma_f32_16x16x32_f16(
            false, a.v, false, b.v, (short)0, acc.v, false, false);
    }
    const int rbase = (lane < 16) ? 0 : 8;
    const int n = (nt << 4) + l15;
#pragma unroll
    for (int r = 0; r < 8; ++r)
        C[(long)((mt << 4) + rbase + r) * N + n] = (_Float16)acc.f[r];
}

// ---------------------------------------------------------------------------
// Gate logit -> monotonic uint key (sigmoid monotonic; top-k set identical)
// ---------------------------------------------------------------------------
__global__ void gate_key_kernel(const float* __restrict__ down,
                                const float* __restrict__ gw,
                                const float* __restrict__ gb,
                                unsigned* __restrict__ keys, int HW) {
    int p = blockIdx.x * blockDim.x + threadIdx.x;
    if (p >= HW) return;
    float s = gb[0];
    for (int c = 0; c < 256; ++c) s += down[(long)c * HW + p] * gw[c];
    unsigned u = __float_as_uint(s);
    u = (u & 0x80000000u) ? ~u : (u | 0x80000000u);
    keys[p] = u;
}

__global__ void hist_kernel(const unsigned* __restrict__ keys,
                            unsigned* __restrict__ hist, int n) {
    int p = blockIdx.x * blockDim.x + threadIdx.x;
    if (p < n) atomicAdd(&hist[keys[p] >> 20], 1u);
}

__global__ void scan_kernel(const unsigned* __restrict__ hist,
                            unsigned* __restrict__ meta, int kk) {
    if (threadIdx.x != 0 || blockIdx.x != 0) return;
    unsigned cum = 0; int thr = 0;
    for (int b = 4095; b >= 0; --b) {
        cum += hist[b];
        if (cum >= (unsigned)kk) { thr = b; break; }
    }
    meta[1] = (unsigned)thr;
}

__global__ void compact_kernel(const unsigned* __restrict__ keys,
                               unsigned* __restrict__ meta,
                               int* __restrict__ idx, int n, int kk) {
    int p = blockIdx.x * blockDim.x + threadIdx.x;
    if (p >= n) return;
    if ((keys[p] >> 20) >= meta[1]) {
        unsigned pos = atomicAdd(&meta[0], 1u);
        if (pos < (unsigned)kk) idx[pos] = p;
    }
}

// ---------------------------------------------------------------------------
// Gather / scatter selected pixels' channel vectors
// ---------------------------------------------------------------------------
__global__ void gather_rows_f32(const float* __restrict__ src,   // C x HW
                                const int* __restrict__ idx,
                                _Float16* __restrict__ dst, int HW) {
    int t = blockIdx.x * blockDim.x + threadIdx.x;
    if (t >= 2048 * 256) return;
    int j = t >> 8, c = t & 255;
    dst[t] = (_Float16)src[(long)c * HW + idx[j]];
}
__global__ void gather_rows_rm(const _Float16* __restrict__ src, // HW x 256
                               const int* __restrict__ idx,
                               _Float16* __restrict__ dst) {
    int t = blockIdx.x * blockDim.x + threadIdx.x;
    if (t >= 2048 * 256) return;
    int j = t >> 8, c = t & 255;
    dst[t] = src[(long)idx[j] * 256 + c];
}

__global__ void scatter_rows(const _Float16* __restrict__ attn,
                             const int* __restrict__ idx,
                             float* __restrict__ out, int HW) {
    int t = blockIdx.x * blockDim.x + threadIdx.x;
    if (t >= 2048 * 256) return;
    int j = t >> 8, c = t & 255;
    out[(long)c * HW + idx[j]] = (float)attn[t];
}

// ---------------------------------------------------------------------------
// Flash attention: 8 heads, hd=32, Nk=2048, one wave per (head, 16 q rows)
// ---------------------------------------------------------------------------
__global__ void __launch_bounds__(256)
flash_attn_wmma(const _Float16* __restrict__ Q,
                const _Float16* __restrict__ Kb,
                const _Float16* __restrict__ Vb,
                _Float16* __restrict__ O, int Nk) {
    __shared__ _Float16 plds[8][16 * 48];
    const int lane = threadIdx.x & 31;
    const int wid  = threadIdx.x >> 5;
    const int gw = blockIdx.x * 8 + wid;   // 0..1023
    const int h  = gw >> 7;                // head
    const int qt = gw & 127;               // q tile
    const int l15 = lane & 15;
    const int aoff  = (lane < 16) ? 0 : 16;
    const int boff  = (lane < 16) ? 0 : 16;
    const int rbase = (lane < 16) ? 0 : 8;

    const _Float16* qp = Q + (long)((qt << 4) + l15) * 256 + h * 32;
    Frag16 qa;
    qa.v8[0] = *(const v8h*)(qp + aoff);
    qa.v8[1] = *(const v8h*)(qp + aoff + 8);

    float rm[8], rs[8];
    FragC o0, o1;
#pragma unroll
    for (int r = 0; r < 8; ++r) { rm[r] = -3.0e38f; rs[r] = 0.0f; o0.f[r] = 0.0f; o1.f[r] = 0.0f; }

    _Float16* pl = plds[wid];
    const float scale = 0.17677669529663687f;  // 1/sqrt(32)

    for (int kt = 0; kt < Nk; kt += 32) {
        Frag16 k0, k1;
        const _Float16* kp0 = Kb + (long)(kt + l15) * 256 + h * 32 + boff;
        const _Float16* kp1 = kp0 + 16 * 256;
        k0.v8[0] = *(const v8h*)(kp0);  k0.v8[1] = *(const v8h*)(kp0 + 8);
        k1.v8[0] = *(const v8h*)(kp1);  k1.v8[1] = *(const v8h*)(kp1 + 8);
        FragC s0, s1;
#pragma unroll
        for (int r = 0; r < 8; ++r) { s0.f[r] = 0.0f; s1.f[r] = 0.0f; }
        s0.v = __builtin_amdgcn_wmma_f32_16x16x32_f16(false, qa.v, false, k0.v, (short)0, s0.v, false, false);
        s1.v = __builtin_amdgcn_wmma_f32_16x16x32_f16(false, qa.v, false, k1.v, (short)0, s1.v, false, false);

#pragma unroll
        for (int r = 0; r < 8; ++r) {
            float a0 = s0.f[r] * scale, a1 = s1.f[r] * scale;
            float tm = fmaxf(a0, a1);
            for (int msk = 1; msk < 16; msk <<= 1) tm = fmaxf(tm, __shfl_xor(tm, msk, 16));
            const float nm = fmaxf(rm[r], tm);
            const float alpha = __expf(rm[r] - nm);
            const float p0 = __expf(a0 - nm);
            const float p1 = __expf(a1 - nm);
            float ps = p0 + p1;
            for (int msk = 1; msk < 16; msk <<= 1) ps += __shfl_xor(ps, msk, 16);
            rs[r] = rs[r] * alpha + ps;
            o0.f[r] *= alpha; o1.f[r] *= alpha;
            rm[r] = nm;
            const int row = rbase + r;
            pl[row * 48 + l15]      = (_Float16)p0;
            pl[row * 48 + 16 + l15] = (_Float16)p1;
        }
        asm volatile("s_wait_dscnt 0" ::: "memory"); // in-wave LDS transpose fence

        Frag16 pf;
        pf.v8[0] = *(const v8h*)(pl + l15 * 48 + aoff);
        pf.v8[1] = *(const v8h*)(pl + l15 * 48 + aoff + 8);
        Frag16 v0, v1;
#pragma unroll
        for (int j = 0; j < 16; ++j) {
            const int key = kt + ((lane < 16) ? 0 : 16) + j;
            const _Float16* vp = Vb + (long)key * 256 + h * 32;
            v0.h[j] = vp[l15];
            v1.h[j] = vp[16 + l15];
        }
        o0.v = __builtin_amdgcn_wmma_f32_16x16x32_f16(false, pf.v, false, v0.v, (short)0, o0.v, false, false);
        o1.v = __builtin_amdgcn_wmma_f32_16x16x32_f16(false, pf.v, false, v1.v, (short)0, o1.v, false, false);
    }

#pragma unroll
    for (int r = 0; r < 8; ++r) {
        const float inv = 1.0f / rs[r];
        const int row = (qt << 4) + rbase + r;
        _Float16* op = O + (long)row * 256 + h * 32;
        op[l15]      = (_Float16)(o0.f[r] * inv);
        op[16 + l15] = (_Float16)(o1.f[r] * inv);
    }
}

// ---------------------------------------------------------------------------
// Host launch
// ---------------------------------------------------------------------------
extern "C" void kernel_launch(void* const* d_in, const int* in_sizes, int n_in,
                              void* d_out, int out_size, void* d_ws, size_t ws_size,
                              hipStream_t stream) {
    (void)in_sizes; (void)n_in; (void)out_size; (void)ws_size;
    const float* down   = (const float*)d_in[0];   // 256 x 256 x 256
    const float* swint  = (const float*)d_in[1];   // 768 x 8 x 8
    const float* wt_w   = (const float*)d_in[2];   // 256 x 768
    const float* wt_b   = (const float*)d_in[3];
    const float* up_dw  = (const float*)d_in[4];   // 5 x 256 x 256 x 4 x 4
    const float* up_db  = (const float*)d_in[5];
    const float* up_cw  = (const float*)d_in[6];   // 5 x 256 x 256 x 3 x 3
    const float* up_cb  = (const float*)d_in[7];
    const float* gate_w = (const float*)d_in[8];
    const float* gate_b = (const float*)d_in[9];
    const float* in_w   = (const float*)d_in[10];  // 768 x 256
    const float* in_b   = (const float*)d_in[11];
    const float* out_w  = (const float*)d_in[12];  // 256 x 256
    const float* out_b  = (const float*)d_in[13];
    const int C = 256, HW = 65536, KSEL = 2048;

    char* wsp = (char*)d_ws;
    auto alloc = [&](size_t bytes) -> void* {
        void* r = (void*)wsp; wsp += (bytes + 255) & ~(size_t)255; return r;
    };
    _Float16* sA      = (_Float16*)alloc((size_t)C * HW * 2);   // pixel-major
    _Float16* sB      = (_Float16*)alloc((size_t)C * HW * 2);
    _Float16* tmp     = (_Float16*)alloc((size_t)C * HW * 2);
    _Float16* swint16 = (_Float16*)alloc((size_t)768 * 64 * 2); // 64 x 768
    _Float16* wtw16   = (_Float16*)alloc((size_t)256 * 768 * 2);
    _Float16* w16     = (_Float16*)alloc((size_t)C * C * 16 * 2);
    _Float16* inw16   = (_Float16*)alloc((size_t)768 * 256 * 2);
    _Float16* outw16  = (_Float16*)alloc((size_t)256 * 256 * 2);
    _Float16* zrow    = (_Float16*)alloc((size_t)768 * 2);      // zero row
    unsigned* keys    = (unsigned*)alloc((size_t)HW * 4);
    unsigned* hist    = (unsigned*)alloc((size_t)(4096 + 8) * 4);
    unsigned* meta    = hist + 4096;               // [0]=counter, [1]=threshold
    int*      idxbuf  = (int*)alloc((size_t)KSEL * 4);
    _Float16* qg      = (_Float16*)alloc((size_t)KSEL * 256 * 2);
    _Float16* kvg     = (_Float16*)alloc((size_t)KSEL * 256 * 2);
    _Float16* qbuf    = (_Float16*)alloc((size_t)KSEL * 256 * 2);
    _Float16* kbuf    = (_Float16*)alloc((size_t)KSEL * 256 * 2);
    _Float16* vbuf    = (_Float16*)alloc((size_t)KSEL * 256 * 2);
    _Float16* obuf    = (_Float16*)alloc((size_t)KSEL * 256 * 2);
    _Float16* attn16  = (_Float16*)alloc((size_t)KSEL * 256 * 2);

    auto cvt = [&](const float* s, _Float16* d, long n) {
        f32_to_f16_kernel<<<(unsigned)((n + 255) / 256), 256, 0, stream>>>(s, d, n);
    };

    // base copy of `down` into output (scatter overwrites selected pixels)
    hipMemcpyAsync(d_out, down, (size_t)C * HW * sizeof(float),
                   hipMemcpyDeviceToDevice, stream);

    // conversions; swint transposed to pixel-major; zero row for borders
    hipMemsetAsync(zrow, 0, (size_t)768 * 2, stream);
    f32_to_f16_transpose_kernel<<<(768 * 64 + 255) / 256, 256, 0, stream>>>(
        swint, swint16, 768, 64);
    cvt(wt_w,  wtw16,  256 * 768);
    cvt(in_w,  inw16,  768 * 256);
    cvt(out_w, outw16, 256 * 256);

    // gate + threshold top-k
    hipMemsetAsync(hist, 0, (4096 + 8) * sizeof(unsigned), stream);
    gate_key_kernel<<<HW / 256, 256, 0, stream>>>(down, gate_w, gate_b, keys, HW);
    hist_kernel<<<HW / 256, 256, 0, stream>>>(keys, hist, HW);
    scan_kernel<<<1, 32, 0, stream>>>(hist, meta, KSEL);
    compact_kernel<<<HW / 256, 256, 0, stream>>>(keys, meta, idxbuf, HW, KSEL);

    auto conv = [&](const _Float16* X, const _Float16* Wt, const float* b, _Float16* Y,
                    int Cin, int Hin, int Ho, int KS, int pad, int dil, int relu) {
        int Mtiles = (Ho * Ho) >> 4, Ntiles = C >> 4;
        int KT = KS * KS * Cin;
        dim3 grid(Ntiles * ((Mtiles + 7) / 8));
        size_t shmem = (size_t)16 * KT * sizeof(_Float16);
        conv_implicit_wmma<<<grid, 256, shmem, stream>>>(
            X, Wt, zrow, b, Y, Cin, Hin, Hin, Ho, Ho, C, KS, pad, dil, relu);
    };

    // 1x1 conv: swint(64x768 pixel-major) -> s(64x256 pixel-major)
    _Float16* cur = sA;
    conv(swint16, wtw16, wt_b, cur, 768, 8, 8, 1, 0, 1, 0);

    // 5x (deconv4x4 s2 + relu -> conv3x3), weights repacked tap-major
    for (int i = 0; i < 5; ++i) {
        int Hi = 8 << i, Ho = Hi * 2;
        pack_tapmajor_kernel<<<(C * C * 16 + 255) / 256, 256, 0, stream>>>(
            up_dw + (long)i * C * C * 16, w16, C, C, 4, /*swap_io=*/1, /*flip=*/1);
        conv(cur, w16, up_db + i * C, tmp, C, Hi, Ho, 4, 2, 2, 1);
        pack_tapmajor_kernel<<<(C * C * 9 + 255) / 256, 256, 0, stream>>>(
            up_cw + (long)i * C * C * 9, w16, C, C, 3, /*swap_io=*/0, /*flip=*/0);
        _Float16* nxt = (cur == sA) ? sB : sA;
        conv(tmp, w16, up_cb + i * C, nxt, C, Ho, Ho, 3, 1, 1, 0);
        cur = nxt;
    }

    // gather selected rows (s is pixel-major: contiguous row copy)
    gather_rows_f32<<<2048, 256, 0, stream>>>(down, idxbuf, qg, HW);
    gather_rows_rm<<<2048, 256, 0, stream>>>(cur, idxbuf, kvg);

    // QKV projections (WMMA GEMM, M=2048 K=256 N=256 each)
    auto gemm = [&](const _Float16* A, const _Float16* B, const float* b,
                    _Float16* Cm, int M, int N, int K) {
        int waves = (M >> 4) * (N >> 4);
        gemm_rcr_wmma<<<(waves + 7) / 8, 256, 0, stream>>>(A, B, b, Cm, M, N, K);
    };
    gemm(qg,  inw16,             in_b,       qbuf, KSEL, 256, 256);
    gemm(kvg, inw16 + 256 * 256, in_b + 256, kbuf, KSEL, 256, 256);
    gemm(kvg, inw16 + 512 * 256, in_b + 512, vbuf, KSEL, 256, 256);

    // attention (8 heads x 128 q-tiles = 1024 waves = 128 blocks)
    flash_attn_wmma<<<128, 256, 0, stream>>>(qbuf, kbuf, vbuf, obuf, KSEL);

    // output projection + scatter
    gemm(obuf, outw16, out_b, attn16, KSEL, 256, 256);
    scatter_rows<<<2048, 256, 0, stream>>>(attn16, idxbuf, (float*)d_out, HW);
}
